// NLBlock_63763084477062
// MI455X (gfx1250) — compile-verified, matrix operand order checked by reading, not run on hardware
//
#include <hip/hip_runtime.h>
#include <hip/hip_bf16.h>
#include <stdint.h>

typedef _Float16 h16;
typedef __attribute__((ext_vector_type(16))) _Float16 v16h;
typedef __attribute__((ext_vector_type(8)))  float    v8f;

#define B_  8
#define C_  1024
#define CI_ 512
#define T_  8
#define H_  28
#define W_  28
#define N_  6272   // T*H*W
#define M_  1568   // T*(H/2)*(W/2)
#define NT_ 49     // M_/32 key tiles
#define SOFT_SCALE 0.04419417382415922f   // 512^-0.5
#define EPS_ 1e-5f

#define QSTR 514   // 512 halves + 1-dword pad (TDM pad_interval=7)
#define VSTR 34    // 32 halves + 1-dword pad (TDM pad_interval=3)
#define SSTR 33
#define PSTR 34

// ---------------------------------------------------------------------------
// WMMA helpers (CDNA5 16x16x32 f16 -> f32)
// ---------------------------------------------------------------------------
static __device__ __forceinline__ v8f wmma32(v16h a, v16h b, v8f c) {
  return __builtin_amdgcn_wmma_f32_16x16x32_f16(false, a, false, b, (short)0, c,
                                                false, false);
}

// 16x32 fragment from LDS tile stored [row][k] (k contiguous), per the CDNA5
// 16-bit A/B VGPR layout (ISA 7.12.2).
static __device__ __forceinline__ v16h ld_frag(const h16* base, int stride_h,
                                               int lane) {
  const int r  = lane & 15;
  const int kb = (lane >> 4) << 3;
  union { v16h v; uint32_t u[8]; } f;
  const h16* p = base + r * stride_h + kb;
#pragma unroll
  for (int i = 0; i < 8; ++i) {
    const int k = ((i & 3) << 1) + ((i >> 2) << 4);
    f.u[i] = *(const uint32_t*)(p + k);
  }
  return f.v;
}

static __device__ __forceinline__ void st_lds_u4(h16* dst, uint4 v) {
  uint32_t* p = (uint32_t*)dst;   // 4B aligned (even half index)
  p[0] = v.x; p[1] = v.y; p[2] = v.z; p[3] = v.w;
}

// ---------------------------------------------------------------------------
// Tensor Data Mover: 2D tile (rows x elems f16) global -> LDS, with hardware
// row padding so the LDS row stride avoids bank conflicts for ld_frag.
// ---------------------------------------------------------------------------
#if __has_builtin(__builtin_amdgcn_tensor_load_to_lds)
#define HAVE_TDM 1
typedef unsigned int u32x4 __attribute__((ext_vector_type(4)));
typedef int          i32x4 __attribute__((ext_vector_type(4)));
typedef int          i32x8 __attribute__((ext_vector_type(8)));

static __device__ __forceinline__ uint32_t lds_off(const void* p) {
  // generic pointers to LDS carry the LDS byte offset in the low 32 bits
  return (uint32_t)(uintptr_t)p;
}

// elems: tile_dim0 == tensor_dim0 (f16 elements per row, row-contiguous)
// tile_rows: tile_dim1; tensor_rows: tensor_dim1 (rows beyond -> zero-fill)
// stride_elems: tensor_dim0_stride; pad_code: pad after 2^(pad_code+1) dwords
static __device__ __forceinline__ void tdm_load_2d(
    uint32_t lds_byte, const void* gp, uint32_t elems, uint32_t tile_rows,
    uint32_t tensor_rows, uint32_t stride_elems, uint32_t pad_code) {
  const uint64_t ga = (uint64_t)(uintptr_t)gp;
  u32x4 g0;
  g0.x = 1u;                                                    // count=1
  g0.y = lds_byte;                                              // lds_addr
  g0.z = (uint32_t)ga;                                          // global lo
  g0.w = (uint32_t)((ga >> 32) & 0x01ffffffu) | 0x80000000u;    // hi | type=2
  i32x8 g1;
  // data_size=2B, pad_enable, pad_interval=pad_code, pad_amount=0 (1 dword)
  g1[0] = (int)((1u << 16) | (1u << 20) | (pad_code << 22));
  g1[1] = (int)((elems & 0xffffu) << 16);                       // td0[15:0]
  g1[2] = (int)(((elems >> 16) & 0xffffu) |
                ((tensor_rows & 0xffffu) << 16));               // td0.hi|td1.lo
  g1[3] = (int)(((tensor_rows >> 16) & 0xffffu) |
                ((elems & 0xffffu) << 16));                     // td1.hi|tile0
  g1[4] = (int)(tile_rows & 0xffffu);                           // tile1|tile2=0
  g1[5] = (int)stride_elems;                                    // stride0 lo32
  g1[6] = 0;
  g1[7] = 0;
  const i32x4 z4 = {0, 0, 0, 0};
#if __clang_major__ >= 23
  const i32x8 z8 = {0, 0, 0, 0, 0, 0, 0, 0};
  __builtin_amdgcn_tensor_load_to_lds(g0, g1, z4, z4, z8, 0);
#else
  __builtin_amdgcn_tensor_load_to_lds(g0, g1, z4, z4, 0);
#endif
}
static __device__ __forceinline__ void tdm_wait0() {
  __builtin_amdgcn_s_wait_tensorcnt(0);
}
#else
#define HAVE_TDM 0
#endif

// ---------------------------------------------------------------------------
// Prep: weights -> f16, BN scale/bias fold
// ---------------------------------------------------------------------------
__global__ void k_prep(const float* __restrict__ tw, const float* __restrict__ pw,
                       const float* __restrict__ gw, const float* __restrict__ ow,
                       const float* __restrict__ bnw, const float* __restrict__ bnb,
                       const float* __restrict__ bnm, const float* __restrict__ bnv,
                       h16* __restrict__ tw16, h16* __restrict__ pw16,
                       h16* __restrict__ gw16, h16* __restrict__ ow16,
                       float* __restrict__ bnsc, float* __restrict__ bnbi) {
  const int i = blockIdx.x * blockDim.x + threadIdx.x;
  if (i < CI_ * C_) {
    tw16[i] = (h16)tw[i];
    pw16[i] = (h16)pw[i];
    gw16[i] = (h16)gw[i];
    ow16[i] = (h16)ow[i];
  }
  if (i < C_) {
    const float s = bnw[i] * rsqrtf(bnv[i] + EPS_);
    bnsc[i] = s;
    bnbi[i] = bnb[i] - bnm[i] * s;
  }
}

// ---------------------------------------------------------------------------
// x [B,C,N] f32  ->  XT [B,N,C] f16
// ---------------------------------------------------------------------------
__global__ __launch_bounds__(256) void k_transpose_x(const float* __restrict__ x,
                                                     h16* __restrict__ XT) {
  __shared__ float tile[32][33];
  const int n0 = blockIdx.x * 32, c0 = blockIdx.y * 32, b = blockIdx.z;
  const int tx = threadIdx.x, ty = threadIdx.y;
#pragma unroll
  for (int i = 0; i < 4; ++i) {
    const int c = c0 + ty + i * 8;
    tile[ty + i * 8][tx] = x[((size_t)(b * C_ + c)) * N_ + n0 + tx];
  }
  __syncthreads();
#pragma unroll
  for (int i = 0; i < 4; ++i) {
    const int n = n0 + ty + i * 8;
    XT[((size_t)b * N_ + n) * C_ + c0 + tx] = (h16)tile[tx][ty + i * 8];
  }
}

// ---------------------------------------------------------------------------
// maxpool(1,2,2) of x, transposed: MPT [B,M,C] f16
// ---------------------------------------------------------------------------
__global__ __launch_bounds__(256) void k_maxpool_t(const float* __restrict__ x,
                                                   h16* __restrict__ MPT) {
  __shared__ float tile[32][33];
  const int m0 = blockIdx.x * 32, c0 = blockIdx.y * 32, b = blockIdx.z;
  const int tx = threadIdx.x, ty = threadIdx.y;
#pragma unroll
  for (int i = 0; i < 4; ++i) {
    const int c = c0 + ty + i * 8;
    const int m = m0 + tx;
    const int tt = m / 196, rem = m % 196, ph = rem / 14, pw = rem % 14;
    const float* xp = x + ((size_t)(b * C_ + c)) * N_ + tt * (H_ * W_) +
                      (2 * ph) * W_ + 2 * pw;
    tile[ty + i * 8][tx] =
        fmaxf(fmaxf(xp[0], xp[1]), fmaxf(xp[W_], xp[W_ + 1]));
  }
  __syncthreads();
#pragma unroll
  for (int i = 0; i < 4; ++i) {
    const int m = m0 + ty + i * 8;
    MPT[((size_t)b * M_ + m) * C_ + c0 + tx] = (h16)tile[tx][ty + i * 8];
  }
}

// ---------------------------------------------------------------------------
// NT GEMM: D[m][n] = sum_k A[m][k]*B[n][k].  128x128 tile, BK=32, 8 waves.
// TDM path: wave0 prefetches tile k+1 by TDM while all waves compute tile k.
// MODE 0: f16 store (+bias_row/+bias_col); MODE 1: f32 BN affine + residual.
// ---------------------------------------------------------------------------
template <int MODE>
__global__ __launch_bounds__(256)
void k_gemm_nt(const h16* __restrict__ Ag, long sA,
               const h16* __restrict__ Bg, long sB,
               void* __restrict__ Dg, long sD,
               int Ma, int Nb, int K,
               const float* __restrict__ bias_row,
               const float* __restrict__ bias_col,
               const float* __restrict__ bnsc,
               const float* __restrict__ bnbi,
               const float* __restrict__ resid, long sR) {
  __shared__ h16 As[2 * 128 * VSTR];
  __shared__ h16 Bs[2 * 128 * VSTR];
  const int b = blockIdx.z;
  const h16* A  = Ag + (size_t)b * sA;
  const h16* Bm = Bg + (size_t)b * sB;
  const int m0 = blockIdx.y * 128, n0 = blockIdx.x * 128;
  const int t = threadIdx.x, lane = t & 31, w = t >> 5;
  const int wm = w >> 2, wn = w & 3;

  v8f acc[4][2];
#pragma unroll
  for (int im = 0; im < 4; ++im)
#pragma unroll
    for (int in_ = 0; in_ < 2; ++in_) acc[im][in_] = (v8f){0, 0, 0, 0, 0, 0, 0, 0};

#if HAVE_TDM
  if (w == 0) {
    tdm_load_2d(lds_off(As), A + (size_t)m0 * K, 32, 128, (uint32_t)(Ma - m0),
                (uint32_t)K, 3);
    tdm_load_2d(lds_off(Bs), Bm + (size_t)n0 * K, 32, 128, (uint32_t)(Nb - n0),
                (uint32_t)K, 3);
    tdm_wait0();
  }
  __syncthreads();
  for (int k0 = 0; k0 < K; k0 += 32) {
    const int cur = (k0 >> 5) & 1;
    if (w == 0 && k0 + 32 < K) {
      const int nxt = cur ^ 1;
      tdm_load_2d(lds_off(As + nxt * 128 * VSTR), A + (size_t)m0 * K + k0 + 32,
                  32, 128, (uint32_t)(Ma - m0), (uint32_t)K, 3);
      tdm_load_2d(lds_off(Bs + nxt * 128 * VSTR), Bm + (size_t)n0 * K + k0 + 32,
                  32, 128, (uint32_t)(Nb - n0), (uint32_t)K, 3);
    }
    const h16* Ac = As + cur * 128 * VSTR;
    const h16* Bc = Bs + cur * 128 * VSTR;
    v16h af[4], bf[2];
#pragma unroll
    for (int im = 0; im < 4; ++im)
      af[im] = ld_frag(Ac + (wm * 64 + im * 16) * VSTR, VSTR, lane);
#pragma unroll
    for (int in_ = 0; in_ < 2; ++in_)
      bf[in_] = ld_frag(Bc + (wn * 32 + in_ * 16) * VSTR, VSTR, lane);
#pragma unroll
    for (int im = 0; im < 4; ++im)
#pragma unroll
      for (int in_ = 0; in_ < 2; ++in_)
        acc[im][in_] = wmma32(af[im], bf[in_], acc[im][in_]);
    if (w == 0) tdm_wait0();
    __syncthreads();
  }
#else
  const int srow = t >> 1, hp = (t & 1) << 4;
  const bool va = (m0 + srow) < Ma;
  const bool vb = (n0 + srow) < Nb;
  const uint4 zz = make_uint4(0, 0, 0, 0);
  for (int k0 = 0; k0 < K; k0 += 32) {
#pragma unroll
    for (int c = 0; c < 2; ++c) {
      const int off = hp + c * 8;
      const uint4 ua = va ? *(const uint4*)(A  + (size_t)(m0 + srow) * K + k0 + off) : zz;
      const uint4 ub = vb ? *(const uint4*)(Bm + (size_t)(n0 + srow) * K + k0 + off) : zz;
      st_lds_u4(As + srow * VSTR + off, ua);
      st_lds_u4(Bs + srow * VSTR + off, ub);
    }
    __syncthreads();
    v16h af[4], bf[2];
#pragma unroll
    for (int im = 0; im < 4; ++im)
      af[im] = ld_frag(As + (wm * 64 + im * 16) * VSTR, VSTR, lane);
#pragma unroll
    for (int in_ = 0; in_ < 2; ++in_)
      bf[in_] = ld_frag(Bs + (wn * 32 + in_ * 16) * VSTR, VSTR, lane);
#pragma unroll
    for (int im = 0; im < 4; ++im)
#pragma unroll
      for (int in_ = 0; in_ < 2; ++in_)
        acc[im][in_] = wmma32(af[im], bf[in_], acc[im][in_]);
    __syncthreads();
  }
#endif

  const int hb = (lane >> 4) << 3, col = lane & 15;
#pragma unroll
  for (int im = 0; im < 4; ++im)
#pragma unroll
    for (int in_ = 0; in_ < 2; ++in_)
#pragma unroll
      for (int j = 0; j < 8; ++j) {
        const int m = m0 + wm * 64 + im * 16 + j + hb;
        const int n = n0 + wn * 32 + in_ * 16 + col;
        if (m < Ma && n < Nb) {
          float v = acc[im][in_][j];
          if (bias_row) v += bias_row[m];
          if (bias_col) v += bias_col[n];
          if (MODE == 0) {
            ((h16*)Dg)[(size_t)b * sD + (size_t)m * Nb + n] = (h16)v;
          } else {
            v = v * bnsc[m] + bnbi[m] + resid[(size_t)b * sR + (size_t)m * Nb + n];
            ((float*)Dg)[(size_t)b * sD + (size_t)m * Nb + n] = v;
          }
        }
      }
}

// ---------------------------------------------------------------------------
// Flash attention: tT[b][n][i] = sum_m softmax_m(theta.phi*scale) * g[i][m]
// 64-query tile resident (Q), keys streamed 32/iter, double-buffered K and V
// staged by the Tensor Data Mover while waves run S-GEMM/softmax/O-GEMM.
// ---------------------------------------------------------------------------
#define SMEM_ATTN ((64 * QSTR + 2 * 32 * QSTR + 2 * 512 * VSTR) * 2 + \
                   64 * SSTR * 4 + 64 * PSTR * 2 + 3 * 64 * 4)

__global__ __launch_bounds__(256)
void k_attn(const h16* __restrict__ ThT, const h16* __restrict__ PhT,
            const h16* __restrict__ G, h16* __restrict__ tT) {
  extern __shared__ char smem[];
  h16*   Qs = (h16*)smem;                 // [64][QSTR]
  h16*   Ks = Qs + 64 * QSTR;             // [2][32][QSTR]
  h16*   Vs = Ks + 2 * 32 * QSTR;         // [2][512][VSTR]
  float* Ss = (float*)(Vs + 2 * 512 * VSTR);  // [64][SSTR]
  h16*   Ps = (h16*)(Ss + 64 * SSTR);     // [64][PSTR]
  float* mrun = (float*)(Ps + 64 * PSTR);
  float* lrun = mrun + 64;
  float* rsc  = lrun + 64;

  const int b  = blockIdx.y;
  const int n0 = blockIdx.x * 64;
  const int t = threadIdx.x, lane = t & 31, w = t >> 5;
  const int hb = (lane >> 4) << 3, col = lane & 15;

  const h16* Qg = ThT + ((size_t)b * N_ + n0) * CI_;
  const h16* Kg = PhT + (size_t)b * M_ * CI_;
  const h16* Gg = G   + (size_t)b * CI_ * M_;

  if (t < 64) { mrun[t] = -3.0e38f; lrun[t] = 0.f; }

  v8f oacc[4][4];
#pragma unroll
  for (int im = 0; im < 4; ++im)
#pragma unroll
    for (int iv = 0; iv < 4; ++iv) oacc[im][iv] = (v8f){0, 0, 0, 0, 0, 0, 0, 0};

#if HAVE_TDM
  if (w == 0) {
    tdm_load_2d(lds_off(Qs), Qg, 512, 64, 64, 512, 7);
    tdm_load_2d(lds_off(Ks), Kg, 512, 32, 32, 512, 7);
    tdm_load_2d(lds_off(Vs), Gg, 32, 512, 512, M_, 3);
    tdm_wait0();
  }
  __syncthreads();
#else
  {  // resident Q tile (sync fallback)
    const int row = t >> 2, q = t & 3;
    const h16* src = Qg + (size_t)row * CI_;
#pragma unroll
    for (int ch = 0; ch < 16; ++ch) {
      const int off = q * 128 + ch * 8;
      st_lds_u4(Qs + row * QSTR + off, *(const uint4*)(src + off));
    }
  }
#endif

  for (int kt = 0; kt < NT_; ++kt) {
    const int mb = kt * 32;
    const int cur = kt & 1;
    const h16* Kc = Ks + cur * 32 * QSTR;
    const h16* Vc = Vs + cur * 512 * VSTR;
#if HAVE_TDM
    if (w == 0 && kt + 1 < NT_) {
      const int nxt = cur ^ 1;
      tdm_load_2d(lds_off(Ks + nxt * 32 * QSTR),
                  Kg + (size_t)(mb + 32) * CI_, 512, 32, 32, 512, 7);
      tdm_load_2d(lds_off(Vs + nxt * 512 * VSTR),
                  Gg + mb + 32, 32, 512, 512, M_, 3);
    }
#else
    {  // stage K tile (32 keys x 512 ch)
      const int row = t >> 3, q = t & 7;
      const h16* src = Kg + (size_t)(mb + row) * CI_;
#pragma unroll
      for (int ch = 0; ch < 8; ++ch) {
        const int off = q * 64 + ch * 8;
        st_lds_u4((h16*)Kc + row * QSTR + off, *(const uint4*)(src + off));
      }
    }
    {  // stage V tile: each wave its own 64(i) x 32(m) slice
#pragma unroll
      for (int rr = 0; rr < 2; ++rr) {
        const int i = lane * 2 + rr;
        const h16* src = Gg + (size_t)(w * 64 + i) * M_ + mb;
#pragma unroll
        for (int ch = 0; ch < 4; ++ch)
          st_lds_u4((h16*)Vc + (w * 64 + i) * VSTR + ch * 8,
                    *(const uint4*)(src + ch * 8));
      }
    }
    __syncthreads();
#endif
    {  // S = Q K^T : each wave one 16x16 of the 64x32 score block
      const int rw = w >> 1, cw = w & 1;
      v8f s = (v8f){0, 0, 0, 0, 0, 0, 0, 0};
#pragma unroll
      for (int ki = 0; ki < 16; ++ki) {
        const v16h aq = ld_frag(Qs + (rw * 16) * QSTR + ki * 32, QSTR, lane);
        const v16h bk = ld_frag(Kc + (cw * 16) * QSTR + ki * 32, QSTR, lane);
        s = wmma32(aq, bk, s);
      }
#pragma unroll
      for (int j = 0; j < 8; ++j)
        Ss[(rw * 16 + j + hb) * SSTR + cw * 16 + col] = s[j] * SOFT_SCALE;
    }
    __syncthreads();
    if (t < 64) {  // online softmax row update
      const float mold = mrun[t];
      float mx = mold;
      float sv[32];
#pragma unroll
      for (int j = 0; j < 32; ++j) { sv[j] = Ss[t * SSTR + j]; mx = fmaxf(mx, sv[j]); }
      const float r = __expf(mold - mx);
      float sum = 0.f;
#pragma unroll
      for (int j = 0; j < 32; ++j) {
        const float p = __expf(sv[j] - mx);
        sum += p;
        Ps[t * PSTR + j] = (h16)p;
      }
      lrun[t] = lrun[t] * r + sum;
      mrun[t] = mx;
      rsc[t] = r;
    }
    __syncthreads();
    {  // O = diag(r) O + P V
      float rf[4][8];
#pragma unroll
      for (int im = 0; im < 4; ++im)
#pragma unroll
        for (int j = 0; j < 8; ++j) rf[im][j] = rsc[im * 16 + j + hb];
      v16h pa[4], vb_[4];
#pragma unroll
      for (int im = 0; im < 4; ++im)
        pa[im] = ld_frag(Ps + (im * 16) * PSTR, PSTR, lane);
#pragma unroll
      for (int iv = 0; iv < 4; ++iv)
        vb_[iv] = ld_frag(Vc + (w * 64 + iv * 16) * VSTR, VSTR, lane);
#pragma unroll
      for (int im = 0; im < 4; ++im)
#pragma unroll
        for (int iv = 0; iv < 4; ++iv) {
#pragma unroll
          for (int j = 0; j < 8; ++j) oacc[im][iv][j] *= rf[im][j];
          oacc[im][iv] = wmma32(pa[im], vb_[iv], oacc[im][iv]);
        }
    }
#if HAVE_TDM
    if (w == 0) tdm_wait0();
#endif
    __syncthreads();
  }

  float il[4][8];
#pragma unroll
  for (int im = 0; im < 4; ++im)
#pragma unroll
    for (int j = 0; j < 8; ++j) il[im][j] = 1.0f / lrun[im * 16 + j + hb];
#pragma unroll
  for (int im = 0; im < 4; ++im)
#pragma unroll
    for (int iv = 0; iv < 4; ++iv)
#pragma unroll
      for (int j = 0; j < 8; ++j) {
        const int n_g = n0 + im * 16 + j + hb;
        const int i_g = w * 64 + iv * 16 + col;
        tT[((size_t)b * N_ + n_g) * CI_ + i_g] = (h16)(oacc[im][iv][j] * il[im][j]);
      }
}

// ---------------------------------------------------------------------------
// Host driver
// ---------------------------------------------------------------------------
extern "C" void kernel_launch(void* const* d_in, const int* in_sizes, int n_in,
                              void* d_out, int out_size, void* d_ws, size_t ws_size,
                              hipStream_t stream) {
  (void)in_sizes; (void)n_in; (void)out_size; (void)ws_size;
  const float* x   = (const float*)d_in[0];
  const float* tw  = (const float*)d_in[1];
  const float* tb  = (const float*)d_in[2];
  const float* pw  = (const float*)d_in[3];
  const float* pb  = (const float*)d_in[4];
  const float* gw  = (const float*)d_in[5];
  const float* gb  = (const float*)d_in[6];
  const float* ow  = (const float*)d_in[7];
  const float* ob  = (const float*)d_in[8];
  const float* bnw = (const float*)d_in[9];
  const float* bnb = (const float*)d_in[10];
  const float* bnm = (const float*)d_in[11];
  const float* bnv = (const float*)d_in[12];

  char* ws = (char*)d_ws;
  size_t off = 0;
  auto alloc = [&](size_t bytes) -> void* {
    void* p = ws + off;
    off += (bytes + 255) & ~(size_t)255;
    return p;
  };
  h16* XT   = (h16*)alloc((size_t)B_ * N_ * C_ * 2);
  h16* MPT  = (h16*)alloc((size_t)B_ * M_ * C_ * 2);
  h16* ThT  = (h16*)alloc((size_t)B_ * N_ * CI_ * 2);
  h16* PhT  = (h16*)alloc((size_t)B_ * M_ * CI_ * 2);
  h16* Gm   = (h16*)alloc((size_t)B_ * CI_ * M_ * 2);
  h16* tT   = (h16*)alloc((size_t)B_ * N_ * CI_ * 2);
  h16* tw16 = (h16*)alloc((size_t)CI_ * C_ * 2);
  h16* pw16 = (h16*)alloc((size_t)CI_ * C_ * 2);
  h16* gw16 = (h16*)alloc((size_t)CI_ * C_ * 2);
  h16* ow16 = (h16*)alloc((size_t)C_ * CI_ * 2);
  float* bnsc = (float*)alloc(C_ * 4);
  float* bnbi = (float*)alloc(C_ * 4);

  const dim3 blk(256);
  k_prep<<<(CI_ * C_ + 255) / 256, blk, 0, stream>>>(
      tw, pw, gw, ow, bnw, bnb, bnm, bnv, tw16, pw16, gw16, ow16, bnsc, bnbi);

  const dim3 b32(32, 8);
  k_transpose_x<<<dim3(N_ / 32, C_ / 32, B_), b32, 0, stream>>>(x, XT);
  k_maxpool_t<<<dim3(M_ / 32, C_ / 32, B_), b32, 0, stream>>>(x, MPT);

  // ThetaT[b][n][i] = XT . theta_w^T (+theta_b)
  k_gemm_nt<0><<<dim3(CI_ / 128, (N_ + 127) / 128, B_), blk, 0, stream>>>(
      XT, (long)N_ * C_, tw16, 0L, ThT, (long)N_ * CI_, N_, CI_, C_,
      nullptr, tb, nullptr, nullptr, nullptr, 0L);
  // PhiT[b][m][i] = MPT . phi_w^T (+phi_b)
  k_gemm_nt<0><<<dim3(CI_ / 128, (M_ + 127) / 128, B_), blk, 0, stream>>>(
      MPT, (long)M_ * C_, pw16, 0L, PhT, (long)M_ * CI_, M_, CI_, C_,
      nullptr, pb, nullptr, nullptr, nullptr, 0L);
  // G[b][i][m] = g_w . MPT^T (+g_b)
  k_gemm_nt<0><<<dim3((M_ + 127) / 128, CI_ / 128, B_), blk, 0, stream>>>(
      gw16, 0L, MPT, (long)M_ * C_, Gm, (long)CI_ * M_, CI_, M_, C_,
      gb, nullptr, nullptr, nullptr, nullptr, 0L);

  // fused flash attention -> tT[b][n][i]
  k_attn<<<dim3(N_ / 64, B_), blk, SMEM_ATTN, stream>>>(ThT, PhT, Gm, tT);

  // out[b][o][n] = (out_w . t + out_b)*bn_scale + bn_bias + x
  k_gemm_nt<1><<<dim3(N_ / 128, C_ / 128, B_), blk, 0, stream>>>(
      ow16, 0L, tT, (long)N_ * CI_, d_out, (long)C_ * N_, C_, N_, CI_,
      ob, nullptr, bnsc, bnbi, x, (long)C_ * N_);
}
